// GSTA_59081570125116
// MI455X (gfx1250) — matrix operand order repaired and checked by reading, not compiled, not run on hardware
//
#include <hip/hip_runtime.h>
#include <hip/hip_bf16.h>

typedef __attribute__((ext_vector_type(16))) _Float16 v16h;
typedef __attribute__((ext_vector_type(8)))  float    v8f;

#define DEVINL __device__ __forceinline__

#if defined(__has_builtin)
#  if __has_builtin(__builtin_amdgcn_tensor_load_to_lds)
#    define HAVE_TDM 1
#  endif
#endif
#ifndef HAVE_TDM
#  define HAVE_TDM 0
#endif

// ---------------------------------------------------------------------------
// WMMA fragment loaders per CDNA5 ISA 7.12.2 (wave32, 16x16x32 f16 -> f32)
// ---------------------------------------------------------------------------

// A (16x32 f16): lanes 0-15 row M=lane hold K 0-7 (e0..7) / 16-23 (e8..15);
// lanes 16-31 row M=lane-16 hold K 8-15 / 24-31.
DEVINL v16h frag_a_rm(const float* __restrict__ A, int row0, int k0, int ld) {
  int l = threadIdx.x & 31; int h = l >> 4; int r = l & 15;
  const float* p = A + (size_t)(row0 + r) * ld + k0;
  v16h a;
#pragma unroll
  for (int e = 0; e < 16; ++e) {
    int k = (e < 8) ? (8 * h + e) : (8 + 8 * h + e);
    a[e] = (_Float16)p[k];
  }
  return a;
}

// A[r][k] = src[(k0+k)*ldk + row0 + r]
DEVINL v16h frag_a_cm(const float* __restrict__ A, int row0, int k0, int ldk) {
  int l = threadIdx.x & 31; int h = l >> 4; int r = l & 15;
  const float* p = A + (size_t)k0 * ldk + row0 + r;
  v16h a;
#pragma unroll
  for (int e = 0; e < 16; ++e) {
    int k = (e < 8) ? (8 * h + e) : (8 + 8 * h + e);
    a[e] = (_Float16)p[(size_t)k * ldk];
  }
  return a;
}

DEVINL v16h frag_a_lds(const _Float16* __restrict__ A, int row0, int k0, int ld) {
  int l = threadIdx.x & 31; int h = l >> 4; int r = l & 15;
  const _Float16* p = A + (size_t)(row0 + r) * ld + k0;
  v16h a;
#pragma unroll
  for (int e = 0; e < 16; ++e) {
    int k = (e < 8) ? (8 * h + e) : (8 + 8 * h + e);
    a[e] = p[k];
  }
  return a;
}

// B (32x16 f16): N = lane%16; lanes 0-15 hold K 0-7 / 16-23, lanes 16-31 K 8-15 / 24-31.
DEVINL v16h frag_b_rm(const float* __restrict__ B, int k0, int col0, int ld) {
  int l = threadIdx.x & 31; int n = l & 15; int kb = (l >> 4) * 8;
  const float* p = B + (size_t)(k0 + kb) * ld + col0 + n;
  v16h b;
#pragma unroll
  for (int e = 0; e < 16; ++e) b[e] = (_Float16)p[(size_t)(e < 8 ? e : e + 8) * ld];
  return b;
}

// B[k][n] = src[(col0+n)*ld + k0 + k]
DEVINL v16h frag_b_cm(const float* __restrict__ Bt, int k0, int col0, int ld) {
  int l = threadIdx.x & 31; int n = l & 15; int kb = (l >> 4) * 8;
  const float* p = Bt + (size_t)(col0 + n) * ld + k0 + kb;
  v16h b;
#pragma unroll
  for (int e = 0; e < 16; ++e) b[e] = (_Float16)p[e < 8 ? e : e + 8];
  return b;
}

DEVINL v8f wmma16(v16h a, v16h b, v8f c) {
  return __builtin_amdgcn_wmma_f32_16x16x32_f16(false, a, false, b, (short)0, c,
                                                false, false);
}

// ---------------------------------------------------------------------------
// Tensor Data Mover descriptors (CDNA5 ISA ch.8, D# groups)
// ---------------------------------------------------------------------------
struct TdmDesc { unsigned int g0[4]; unsigned int g1[8]; };

DEVINL void set_bits(unsigned int* d, int lo, int len, unsigned long long v) {
#pragma unroll
  for (int i = 0; i < 64; ++i) {
    if (i >= len) break;
    int bit = lo + i;
    unsigned int m = 1u << (bit & 31);
    if ((v >> i) & 1ull) d[bit >> 5] |= m;
  }
}

// Exact-fit 2D tile: tensor dims == tile dims (no OOB), data_size = 4 bytes.
DEVINL void tdm_build(TdmDesc& d, unsigned int lds_off, unsigned int tile0,
                      unsigned int tile1, unsigned long long stride0_elems) {
#pragma unroll
  for (int i = 0; i < 4; ++i) d.g0[i] = 0u;
#pragma unroll
  for (int i = 0; i < 8; ++i) d.g1[i] = 0u;
  set_bits(d.g0, 0, 2, 1);                 // count = 1 (valid descriptor)
  set_bits(d.g0, 32, 32, lds_off);         // lds_addr (bytes)
  set_bits(d.g0, 126, 2, 2);               // type = 2 ("image")
  set_bits(d.g1, 16, 2, 2);                // data_size = 4 bytes
  set_bits(d.g1, 48, 32, tile0);           // tensor_dim0
  set_bits(d.g1, 80, 32, tile1);           // tensor_dim1
  set_bits(d.g1, 112, 16, tile0);          // tile_dim0
  set_bits(d.g1, 128, 16, tile1);          // tile_dim1
  set_bits(d.g1, 160, 48, stride0_elems);  // tensor_dim0_stride (elements)
}

DEVINL void tdm_set_gaddr(TdmDesc& d, unsigned long long gaddr) {
  d.g0[2] = (unsigned int)gaddr;
  d.g0[3] = (unsigned int)((gaddr >> 32) & 0x01FFFFFFull) | (2u << 30);
}

#if HAVE_TDM
typedef unsigned int u32x4 __attribute__((ext_vector_type(4)));
typedef int          i32x8 __attribute__((ext_vector_type(8)));
typedef int          i32x4 __attribute__((ext_vector_type(4)));

DEVINL void tdm_issue(const TdmDesc& d) {
  u32x4 g0; i32x8 g1;
  i32x4 z4 = {0, 0, 0, 0};
  i32x8 z8 = {0, 0, 0, 0, 0, 0, 0, 0};
#pragma unroll
  for (int i = 0; i < 4; ++i) g0[i] = d.g0[i];
#pragma unroll
  for (int i = 0; i < 8; ++i) g1[i] = (int)d.g1[i];
  // clang-23 / amdgpu-toolchain form: 6 args (g0, g1, g2, g3, g4, cpol)
  __builtin_amdgcn_tensor_load_to_lds(g0, g1, z4, z4, z8, 0);
}
#endif

// ---------------------------------------------------------------------------
// TDM-staged workgroup GEMM: 128 threads (4 waves), 64x64 output tile.
// out[z][row][col] = sum_k A[z][row][k] * B(k,col) (+bias[row])
//   A: row-major (lda), B: bColMajor ? B[col][k] (ldb) : B[k][col] (ldb)
// Per K-step of 32: wave0 TDM-loads A slab (64x32) and B slab into LDS,
// s_wait_tensorcnt 0, workgroup barrier, 4 WMMAs per wave from LDS.
// ---------------------------------------------------------------------------
__global__ void k_gemm_tdm(const float* __restrict__ A, size_t aStrideZ, int lda,
                           const float* __restrict__ B, size_t bStrideZ, int ldb,
                           int bColMajor, const float* __restrict__ bias,
                           float* __restrict__ out, size_t oStrideZ, int ldo,
                           int Kdim) {
  __shared__ float smem[64 * 32 + 64 * 32];   // A slab (8KB) + B slab (8KB)
  float* As = smem;
  float* Bs = smem + 64 * 32;

  int col0 = blockIdx.x * 64, row0 = blockIdx.y * 64, z = blockIdx.z;
  const float* Az = A + (size_t)z * aStrideZ;
  const float* Bz = B + (size_t)z * bStrideZ;
  int wave = threadIdx.x >> 5;
  int mw = (wave & 1) * 32, nw = (wave >> 1) * 32;

  TdmDesc dA, dB;
  tdm_build(dA, 0u, 32u, 64u, (unsigned long long)lda);
  if (bColMajor) tdm_build(dB, 64u * 32u * 4u, 32u, 64u, (unsigned long long)ldb);
  else           tdm_build(dB, 64u * 32u * 4u, 64u, 32u, (unsigned long long)ldb);

  v8f acc[2][2] = {};

  for (int k0 = 0; k0 < Kdim; k0 += 32) {
#if HAVE_TDM
    if (wave == 0) {
      unsigned long long aAddr =
          (unsigned long long)(Az + (size_t)row0 * lda + k0);
      unsigned long long bAddr = bColMajor
          ? (unsigned long long)(Bz + (size_t)col0 * ldb + k0)
          : (unsigned long long)(Bz + (size_t)k0 * ldb + col0);
      tdm_set_gaddr(dA, aAddr);
      tdm_set_gaddr(dB, bAddr);
      tdm_issue(dA);
      tdm_issue(dB);
      __builtin_amdgcn_s_wait_tensorcnt(0);
    }
#else
    // cooperative fallback staging
    for (int i = threadIdx.x; i < 64 * 32; i += 128) {
      int r = i >> 5, c = i & 31;
      As[i] = Az[(size_t)(row0 + r) * lda + k0 + c];
      if (bColMajor) Bs[i] = Bz[(size_t)(col0 + r) * ldb + k0 + c];
      else {
        int kr = i >> 6, cc = i & 63;
        Bs[i] = Bz[(size_t)(k0 + kr) * ldb + col0 + cc];
      }
    }
#endif
    __syncthreads();

    v16h a0 = frag_a_rm(As, mw, 0, 32);
    v16h a1 = frag_a_rm(As, mw + 16, 0, 32);
    v16h b0, b1;
    if (bColMajor) {
      b0 = frag_b_cm(Bs, 0, nw, 32);
      b1 = frag_b_cm(Bs, 0, nw + 16, 32);
    } else {
      b0 = frag_b_rm(Bs, 0, nw, 64);
      b1 = frag_b_rm(Bs, 0, nw + 16, 64);
    }
    acc[0][0] = wmma16(a0, b0, acc[0][0]);
    acc[0][1] = wmma16(a0, b1, acc[0][1]);
    acc[1][0] = wmma16(a1, b0, acc[1][0]);
    acc[1][1] = wmma16(a1, b1, acc[1][1]);
    __syncthreads();
  }

  int l = threadIdx.x & 31; int n = l & 15; int mo = (l >> 4) * 8;
  float* oz = out + (size_t)z * oStrideZ;
#pragma unroll
  for (int i = 0; i < 2; ++i)
#pragma unroll
    for (int j = 0; j < 2; ++j)
#pragma unroll
      for (int r = 0; r < 8; ++r) {
        int row = row0 + mw + 16 * i + mo + r;
        int col = col0 + nw + 16 * j + n;
        float v = acc[i][j][r];
        if (bias) v += bias[row];
        oz[(size_t)row * ldo + col] = v;
      }
}

// ---------------------------------------------------------------------------
// Small-K attention kernels (wave-per-tile)
// ---------------------------------------------------------------------------

// E[b][m][n] = sum_k q[b][k][m] * kk[b][k][n]
__global__ void k_pam_energy(const float* __restrict__ Q, const float* __restrict__ Kc,
                             float* __restrict__ E, int C8, int M) {
  int n0 = blockIdx.x * 16, m0 = blockIdx.y * 16, b = blockIdx.z;
  const float* qb = Q + (size_t)b * C8 * M;
  const float* kb = Kc + (size_t)b * C8 * M;
  v8f acc = {};
  for (int k0 = 0; k0 < C8; k0 += 32) {
    v16h a = frag_a_cm(qb, m0, k0, M);
    v16h bf = frag_b_rm(kb, k0, n0, M);
    acc = wmma16(a, bf, acc);
  }
  int l = threadIdx.x & 31; int n = l & 15; int mo = (l >> 4) * 8;
  float* eb = E + (size_t)b * M * M;
#pragma unroll
  for (int r = 0; r < 8; ++r)
    eb[(size_t)(m0 + mo + r) * M + n0 + n] = acc[r];
}

__global__ void k_softmax128(float* __restrict__ E) {
  __shared__ float red[128];
  float* p = E + (size_t)blockIdx.x * 128;
  int t = threadIdx.x;
  float v = p[t];
  red[t] = v; __syncthreads();
  for (int s = 64; s > 0; s >>= 1) { if (t < s) red[t] = fmaxf(red[t], red[t + s]); __syncthreads(); }
  float mx = red[0]; __syncthreads();
  float e = __expf(v - mx);
  red[t] = e; __syncthreads();
  for (int s = 64; s > 0; s >>= 1) { if (t < s) red[t] += red[t + s]; __syncthreads(); }
  p[t] = e / red[0];
}

// Y[b][c][m] = gamma * sum_n V[b][c][n]*ATT[b][m][n] + X[b][c][m]
__global__ void k_pam_out(const float* __restrict__ Vm, const float* __restrict__ ATT,
                          const float* __restrict__ X, const float* __restrict__ gptr,
                          float* __restrict__ Y, int C, int M) {
  int m0 = blockIdx.x * 16, c0 = blockIdx.y * 16, b = blockIdx.z;
  const float* vb = Vm + (size_t)b * C * M;
  const float* ab = ATT + (size_t)b * M * M;
  float g = gptr[0];
  v8f acc = {};
  for (int k0 = 0; k0 < M; k0 += 32) {
    v16h a = frag_a_rm(vb, c0, k0, M);
    v16h bf = frag_b_cm(ab, k0, m0, M);
    acc = wmma16(a, bf, acc);
  }
  int l = threadIdx.x & 31; int n = l & 15; int mo = (l >> 4) * 8;
  const float* xb = X + (size_t)b * C * M;
  float* yb = Y + (size_t)b * C * M;
#pragma unroll
  for (int r = 0; r < 8; ++r) {
    size_t idx = (size_t)(c0 + mo + r) * M + m0 + n;
    yb[idx] = g * acc[r] + xb[idx];
  }
}

// CAM fused: 16 channel-rows per block; 16x2048 energy strip in LDS (f16).
// softmax(rowmax - E) == exp(rowmin - E)/sum.
__global__ void k_cam(const float* __restrict__ Y, const float* __restrict__ gptr,
                      float* __restrict__ OUT, int C, int M) {
  __shared__ _Float16 Esh[16 * 2048];
  __shared__ float redA[16][8];
  __shared__ float redB[16][8];
  int b = blockIdx.y;
  int c0 = blockIdx.x * 16;
  const float* yb = Y + (size_t)b * C * M;
  int wave = threadIdx.x >> 5;
  int l = threadIdx.x & 31; int n = l & 15; int mo = (l >> 4) * 8;

  for (int dt = wave; dt < C / 16; dt += 4) {
    int d0 = dt * 16;
    v8f acc = {};
    for (int k0 = 0; k0 < M; k0 += 32) {
      v16h a = frag_a_rm(yb, c0, k0, M);
      v16h bf = frag_b_cm(yb, k0, d0, M);
      acc = wmma16(a, bf, acc);
    }
#pragma unroll
    for (int r = 0; r < 8; ++r)
      Esh[(mo + r) * C + d0 + n] = (_Float16)acc[r];
  }
  __syncthreads();

  int rrow = threadIdx.x >> 3, sub = threadIdx.x & 7;
  float mn = 1e30f;
  for (int d = sub; d < C; d += 8) mn = fminf(mn, (float)Esh[rrow * C + d]);
  redA[rrow][sub] = mn; __syncthreads();
  if (sub == 0) {
    float m2 = redA[rrow][0];
#pragma unroll
    for (int i = 1; i < 8; ++i) m2 = fminf(m2, redA[rrow][i]);
    redA[rrow][0] = m2;
  }
  __syncthreads();
  mn = redA[rrow][0];
  float s = 0.f;
  for (int d = sub; d < C; d += 8) {
    float e = __expf(mn - (float)Esh[rrow * C + d]);
    Esh[rrow * C + d] = (_Float16)e;
    s += e;
  }
  redB[rrow][sub] = s; __syncthreads();
  if (sub == 0) {
    float s2 = 0.f;
#pragma unroll
    for (int i = 0; i < 8; ++i) s2 += redB[rrow][i];
    redB[rrow][0] = s2;
  }
  __syncthreads();
  float inv = 1.f / redB[rrow][0];
  for (int d = sub; d < C; d += 8)
    Esh[rrow * C + d] = (_Float16)((float)Esh[rrow * C + d] * inv);
  __syncthreads();

  float g = gptr[0];
  float* ob = OUT + (size_t)b * C * M;
  for (int mt = wave; mt < M / 16; mt += 4) {
    int m0 = mt * 16;
    v8f acc = {};
    for (int k0 = 0; k0 < C; k0 += 32) {
      v16h a = frag_a_lds(Esh, 0, k0, C);
      v16h bf = frag_b_rm(yb, k0, m0, M);
      acc = wmma16(a, bf, acc);
    }
#pragma unroll
    for (int r = 0; r < 8; ++r) {
      size_t idx = (size_t)(c0 + mo + r) * M + m0 + n;
      ob[idx] = g * acc[r] + yb[idx];
    }
  }
}

// ---------------------------------------------------------------------------
// Graph-layer helpers
// ---------------------------------------------------------------------------

// graph[n][v][w] = 0.5*(l1norm(mask*adj) + l1norm(mask*sim)) per row v
__global__ void k_graph_build(const float* __restrict__ GR, const float* __restrict__ ADJ,
                              float* __restrict__ G, int V) {
  __shared__ float red[128];
  int v = blockIdx.x, nb = blockIdx.y, w = threadIdx.x;
  const float* grb = GR + (size_t)nb * V * V;
  const float* ab = ADJ + (size_t)nb * V * V;
  float sqv = grb[(size_t)v * V + v];
  float sqw = grb[(size_t)w * V + w];
  float gvw = grb[(size_t)v * V + w];
  float mask = (w == v) ? 0.f : 1.f;
  float d2 = fmaxf(sqv + sqw - 2.f * gvw, 1e-12f);
  float dist = sqrtf(d2);
  float sim = mask * (2.f / (__expf(dist) + 1.f));
  float a = mask * ab[(size_t)v * V + w];

  red[w] = fabsf(a); __syncthreads();
  for (int s = 64; s > 0; s >>= 1) { if (w < s) red[w] += red[w + s]; __syncthreads(); }
  float sa = fmaxf(red[0], 1e-12f); __syncthreads();
  red[w] = fabsf(sim); __syncthreads();
  for (int s = 64; s > 0; s >>= 1) { if (w < s) red[w] += red[w + s]; __syncthreads(); }
  float ss = fmaxf(red[0], 1e-12f);

  G[(size_t)nb * V * V + (size_t)v * V + w] = 0.5f * (a / sa + sim / ss);
}

// HP[n][v][o] = sum_w graph[n][v][w] * H[n][w][o]   (K = 128)
__global__ void k_gemm_hp(const float* __restrict__ G, const float* __restrict__ H,
                          float* __restrict__ HP, int V, int C) {
  int o0 = blockIdx.x * 16, v0 = blockIdx.y * 16, nb = blockIdx.z;
  const float* gb = G + (size_t)nb * V * V;
  const float* hb = H + (size_t)nb * V * C;
  v8f acc = {};
  for (int k0 = 0; k0 < V; k0 += 32) {
    v16h a = frag_a_rm(gb, v0, k0, V);
    v16h bf = frag_b_rm(hb, k0, o0, C);
    acc = wmma16(a, bf, acc);
  }
  int l = threadIdx.x & 31; int n = l & 15; int mo = (l >> 4) * 8;
  float* ob = HP + (size_t)nb * V * C;
#pragma unroll
  for (int r = 0; r < 8; ++r)
    ob[(size_t)(v0 + mo + r) * C + o0 + n] = acc[r];
}

__global__ void k_bn_stats(const float* __restrict__ HP, float* __restrict__ mean,
                           float* __restrict__ var, int rows, int C) {
  int o = blockIdx.x * blockDim.x + threadIdx.x;
  if (o >= C) return;
  float s = 0.f, ss = 0.f;
  for (int r = 0; r < rows; ++r) {
    float v = HP[(size_t)r * C + o];
    s += v; ss += v * v;
  }
  float m = s / (float)rows;
  mean[o] = m;
  var[o] = ss / (float)rows - m * m;
}

__global__ void k_bn_apply(const float* __restrict__ HP, const float* __restrict__ inp,
                           const float* __restrict__ mean, const float* __restrict__ var,
                           const float* __restrict__ bw, const float* __restrict__ bb,
                           const float* __restrict__ gptr, int gidx,
                           float* __restrict__ out, int total, int C) {
  int i = blockIdx.x * blockDim.x + threadIdx.x;
  if (i >= total) return;
  int o = i % C;
  float hb = (HP[i] - mean[o]) * rsqrtf(var[o] + 1e-5f) * bw[o] + bb[o];
  float act = hb > 0.f ? hb : 0.1f * hb;
  out[i] = inp[i] + gptr[gidx] * act;
}

// ---------------------------------------------------------------------------
// Host launcher
// ---------------------------------------------------------------------------
extern "C" void kernel_launch(void* const* d_in, const int* in_sizes, int n_in,
                              void* d_out, int out_size, void* d_ws, size_t ws_size,
                              hipStream_t stream) {
  const float* x     = (const float*)d_in[0];
  const float* vfeat = (const float*)d_in[1];
  const float* adj   = (const float*)d_in[2];
  const float* Wq    = (const float*)d_in[3];
  const float* bq    = (const float*)d_in[4];
  const float* Wk    = (const float*)d_in[5];
  const float* bk    = (const float*)d_in[6];
  const float* Wv    = (const float*)d_in[7];
  const float* bv    = (const float*)d_in[8];
  const float* gpam  = (const float*)d_in[9];
  const float* gcam  = (const float*)d_in[10];
  const float* Wg    = (const float*)d_in[11];
  const float* bnw   = (const float*)d_in[12];
  const float* bnb   = (const float*)d_in[13];
  const float* gg    = (const float*)d_in[14];

  const int B = 16, C = 2048, HW = 128, C8 = 256, N = 16, V = 128, G = 2;

  float* out_y = (float*)d_out;
  float* out_g = out_y + (size_t)B * C * HW;

  float* ws   = (float*)d_ws;
  float* q    = ws;
  float* kq   = q    + (size_t)B * C8 * HW;
  float* vv   = kq   + (size_t)B * C8 * HW;
  float* att  = vv   + (size_t)B * C  * HW;
  float* ypam = att  + (size_t)B * HW * HW;

  float* h    = ws;                             // reuse after CAM completes
  float* gram = h    + (size_t)N * V * C;
  float* gw   = gram + (size_t)N * V * V;
  float* hp   = gw   + (size_t)N * V * V;
  float* gbuf = hp   + (size_t)N * V * C;
  float* mean = gbuf + (size_t)N * V * C;
  float* var  = mean + C;

  dim3 w32(32), w128(128);

  // ---- PAM projections: TDM-staged workgroup GEMMs ----
  // q/k: (C8 x C) x (C x HW) per batch; v: (C x C) x (C x HW)
  k_gemm_tdm<<<dim3(HW / 64, C8 / 64, B), w128, 0, stream>>>(
      Wq, 0, C, x, (size_t)C * HW, HW, 0, bq, q, (size_t)C8 * HW, HW, C);
  k_gemm_tdm<<<dim3(HW / 64, C8 / 64, B), w128, 0, stream>>>(
      Wk, 0, C, x, (size_t)C * HW, HW, 0, bk, kq, (size_t)C8 * HW, HW, C);
  k_gemm_tdm<<<dim3(HW / 64, C / 64, B), w128, 0, stream>>>(
      Wv, 0, C, x, (size_t)C * HW, HW, 0, bv, vv, (size_t)C * HW, HW, C);

  k_pam_energy<<<dim3(HW / 16, HW / 16, B), w32, 0, stream>>>(q, kq, att, C8, HW);
  k_softmax128<<<dim3(B * HW), w128, 0, stream>>>(att);
  k_pam_out<<<dim3(HW / 16, C / 16, B), w32, 0, stream>>>(vv, att, x, gpam, ypam, C, HW);

  // ---- CAM (fused, writes final y) ----
  k_cam<<<dim3(C / 16, B), w128, 0, stream>>>(ypam, gcam, out_y, C, HW);

  // ---- Graph layers ----
  const float* gin = vfeat;
  for (int i = 0; i < G; ++i) {
    const float* Wl = Wg + (size_t)i * C * C;
    // h[n] = inp[n] (VxC) x Wl^T (CxC): B col-major (Wl rows are k-contiguous)
    k_gemm_tdm<<<dim3(C / 64, V / 64, N), w128, 0, stream>>>(
        gin, (size_t)V * C, C, Wl, 0, C, 1, nullptr, h, (size_t)V * C, C, C);
    // gram[n] = inp[n] x inp[n]^T
    k_gemm_tdm<<<dim3(V / 64, V / 64, N), w128, 0, stream>>>(
        gin, (size_t)V * C, C, gin, (size_t)V * C, C, 1, nullptr,
        gram, (size_t)V * V, V, C);
    k_graph_build<<<dim3(V, N), w128, 0, stream>>>(gram, adj, gw, V);
    k_gemm_hp<<<dim3(C / 16, V / 16, N), w32, 0, stream>>>(gw, h, hp, V, C);
    k_bn_stats<<<dim3(C / 256), dim3(256), 0, stream>>>(hp, mean, var, N * V, C);
    float* dst = (i == G - 1) ? out_g : gbuf;
    k_bn_apply<<<dim3((N * V * C) / 256), dim3(256), 0, stream>>>(
        hp, gin, mean, var, bnw + (size_t)i * C, bnb + (size_t)i * C,
        gg, i, dst, N * V * C, C);
    gin = gbuf;
  }
}
